// LSTMfromScratch_58025008169376
// MI455X (gfx1250) — matrix operand order repaired and checked by reading, not compiled, not run on hardware
//
#include <hip/hip_runtime.h>

// ---------------------------------------------------------------------------
// 2-layer LSTM (c == h quirk) + final Linear, for MI455X (gfx1250, wave32).
//
// Strategy (see analysis): persistent-grid kernel, 64 workgroups x 128 thr.
// Each WG owns 4 hidden units (16 gate columns) for BOTH layers; its bf16
// weight slices live in LDS for the whole T=2048 recurrence. Per step:
//   z0 = h0 @ Wh0^T + x_t @ Wx0^T      (WMMA bf16 16x16x32, f32 accum)
//   h0' = cell(z0)                      -> global bf16 (double buffered)
//   [global barrier]
//   z1 = h1 @ Wh1^T + h0' @ Wx1^T
//   h1' = cell(z1)
//   [global barrier]
// Final step also writes hidden_out (f32) and 'last'; then each WG computes
// its 4 columns of the FC output. 2 spin barriers/step, weights never leave
// LDS, x/h traffic is all L2-resident (192MB L2 >> 128MB x + 4MB W).
// ---------------------------------------------------------------------------

typedef __bf16 bf16;
typedef __attribute__((ext_vector_type(16))) __bf16 bvec16;
typedef __attribute__((ext_vector_type(8)))  __bf16 bvec8;
typedef __attribute__((ext_vector_type(8)))  float  fvec8;

#define B_N   64
#define T_N   2048
#define H_N   256
#define I_N   256
#define KW    (I_N + H_N)      // 512: weight row length (h part then x part)
#define NWG   64               // persistent workgroups
#define UPW   4                // hidden units per WG
#define COLS  16               // 4 gates * UPW gate-columns per WG
#define NTHR  128              // 4 waves -> 4 C tiles of 16x16 (M=64, N=16)

struct Params {
  const float* x;        // [B, T, I]
  const float* hidden;   // [2, B, H]
  const float* fc_w;     // [H, H]
  const float* fc_b;     // [H]
  const float* W[8];     // [layer*4 + gate] -> [H, KW], gates f,i,g,o
  const float* bias[8];  // [layer*4 + gate] -> [H]
  bf16*  h0bf[2];        // [B, H] bf16, double buffered
  bf16*  h1bf[2];
  float* last;           // [B, H] f32 final layer-1 state
  float* out;            // d_out: [B*H output][2*B*H hidden_out]
  unsigned* bar;         // global barrier counter
};

__device__ __forceinline__ bvec16 cat8(bvec8 a, bvec8 b) {
  return __builtin_shufflevector(a, b, 0,1,2,3,4,5,6,7,8,9,10,11,12,13,14,15);
}

__device__ __forceinline__ float sigf(float x) {
  return 1.0f / (1.0f + __expf(-x));
}

// A fragment (16x32 bf16, row-major source, row stride = H_N) per ISA layout:
// lane<16: K kb+0..7 (v0..3), kb+16..23 (v4..7); lane>=16: +8.
__device__ __forceinline__ fvec8 gemm_h(const bf16* __restrict__ Abase, int mtile,
                                        const bf16* __restrict__ Wls, int lane,
                                        fvec8 c) {
#pragma unroll
  for (int kb = 0; kb < H_N; kb += 32) {
    const int r  = mtile * 16 + (lane & 15);
    const int ka = kb + ((lane & 16) ? 8 : 0);
    const bf16* pa = Abase + r * H_N + ka;
    const bvec16 a = cat8(*(const bvec8*)pa, *(const bvec8*)(pa + 16));
    const int col = lane & 15;
    const int kbb = kb + ((lane & 16) ? 8 : 0);
    const bf16* pb = Wls + col * H_N + kbb;
    const bvec16 b = cat8(*(const bvec8*)pb, *(const bvec8*)(pb + 16));
    c = __builtin_amdgcn_wmma_f32_16x16x32_bf16(false, a, false, b,
                                                (short)0, c, false, false);
  }
  return c;
}

// Same, but A rows come from f32 x[b][t][:] (row stride T_N*I_N), converted
// to bf16 in-register (VALU converts co-execute with the XDL wmma pipe).
__device__ __forceinline__ fvec8 gemm_x(const float* __restrict__ xbase, int mtile,
                                        const bf16* __restrict__ Wls, int lane,
                                        fvec8 c) {
#pragma unroll
  for (int kb = 0; kb < I_N; kb += 32) {
    const int r  = mtile * 16 + (lane & 15);
    const int ka = kb + ((lane & 16) ? 8 : 0);
    const float* pa = xbase + (size_t)r * (T_N * I_N) + ka;
    const fvec8 f0 = *(const fvec8*)pa;
    const fvec8 f1 = *(const fvec8*)(pa + 16);
    bvec8 a0, a1;
#pragma unroll
    for (int e = 0; e < 8; ++e) { a0[e] = (bf16)f0[e]; a1[e] = (bf16)f1[e]; }
    const bvec16 a = cat8(a0, a1);
    const int col = lane & 15;
    const int kbb = kb + ((lane & 16) ? 8 : 0);
    const bf16* pb = Wls + col * H_N + kbb;
    const bvec16 b = cat8(*(const bvec8*)pb, *(const bvec8*)(pb + 16));
    c = __builtin_amdgcn_wmma_f32_16x16x32_bf16(false, a, false, b,
                                                (short)0, c, false, false);
  }
  return c;
}

// Grid-wide spin barrier: release arrival, acquire poll (s_sleep between
// polls). All 64 WGs are trivially co-resident (128 thr / ~37KB LDS each).
__device__ __forceinline__ void gbar(unsigned* bar, unsigned target) {
  __syncthreads();
  if (threadIdx.x == 0) {
    __hip_atomic_fetch_add(bar, 1u, __ATOMIC_RELEASE, __HIP_MEMORY_SCOPE_AGENT);
    while (__hip_atomic_load(bar, __ATOMIC_ACQUIRE, __HIP_MEMORY_SCOPE_AGENT)
           < target) {
      __builtin_amdgcn_s_sleep(1);
    }
  }
  __syncthreads();
}

__global__ void init_kernel(const float* __restrict__ hidden,
                            bf16* __restrict__ h0, bf16* __restrict__ h1,
                            unsigned* __restrict__ bar) {
  const int i = blockIdx.x * blockDim.x + threadIdx.x;
  if (i == 0) *bar = 0u;
  if (i < B_N * H_N) {
    h0[i] = (bf16)hidden[i];
    h1[i] = (bf16)hidden[B_N * H_N + i];
  }
}

__global__ __launch_bounds__(NTHR, 1) void lstm_kernel(Params p) {
  __shared__ __align__(32) bf16 Wh0s[COLS * H_N];
  __shared__ __align__(32) bf16 Wx0s[COLS * H_N];
  __shared__ __align__(32) bf16 Wh1s[COLS * H_N];
  __shared__ __align__(32) bf16 Wx1s[COLS * H_N];
  __shared__ float zbuf[B_N * COLS];
  __shared__ float bias0s[COLS], bias1s[COLS];

  const int wg   = blockIdx.x;
  const int tid  = threadIdx.x;
  const int lane = tid & 31;
  const int wv   = tid >> 5;          // wave id == M tile (batch rows 16*wv..)

  // --- Stage this WG's weight slices into LDS (f32 -> bf16), col-major-by-K:
  // column c = gate*UPW + u  <->  hidden unit j = wg*UPW + u of gate.
  for (int idx = tid; idx < COLS * H_N; idx += NTHR) {
    const int c = idx >> 8, k = idx & 255;
    const int gate = c >> 2, u = c & 3;
    const size_t row = (size_t)(wg * UPW + u) * KW;
    Wh0s[idx] = (bf16)p.W[gate    ][row + k];
    Wx0s[idx] = (bf16)p.W[gate    ][row + H_N + k];
    Wh1s[idx] = (bf16)p.W[4 + gate][row + k];
    Wx1s[idx] = (bf16)p.W[4 + gate][row + H_N + k];
  }
  if (tid < COLS) {
    const int gate = tid >> 2, u = tid & 3;
    bias0s[tid] = p.bias[gate    ][wg * UPW + u];
    bias1s[tid] = p.bias[4 + gate][wg * UPW + u];
  }

  // Per-thread recurrent state: 2 (batch,unit) pairs per layer, in registers.
  float hp0[2], hp1[2];
#pragma unroll
  for (int q = 0; q < 2; ++q) {
    const int pp = tid * 2 + q;
    const int b = pp >> 2, u = pp & 3;
    const int j = wg * UPW + u;
    hp0[q] = p.hidden[(size_t)b * H_N + j];
    hp1[q] = p.hidden[(size_t)(B_N * H_N) + b * H_N + j];
  }
  __syncthreads();

  unsigned epoch = 0;
  for (int t = 0; t < T_N; ++t) {
    const int cur = t & 1, nxt = cur ^ 1;

    // ---------------- Layer 0 gates: z0 = h0@Wh0^T + x_t@Wx0^T ------------
    fvec8 ch = {};
    ch = gemm_h(p.h0bf[cur], wv, Wh0s, lane, ch);
    fvec8 cx = {};
    cx = gemm_x(p.x + (size_t)t * I_N, wv, Wx0s, lane, cx);
    {
      const fvec8 z = ch + cx;
      const int colz = lane & 15;
      const int rowb = wv * 16 + ((lane & 16) ? 8 : 0);
#pragma unroll
      for (int r = 0; r < 8; ++r) zbuf[(rowb + r) * COLS + colz] = z[r];
    }
    __syncthreads();

    // cell 0
#pragma unroll
    for (int q = 0; q < 2; ++q) {
      const int pp = tid * 2 + q;
      const int b = pp >> 2, u = pp & 3;
      const float f = sigf(zbuf[b * COLS +      u] + bias0s[u]);
      const float i = sigf(zbuf[b * COLS +  4 + u] + bias0s[4 + u]);
      const float g = tanhf(zbuf[b * COLS +  8 + u] + bias0s[8 + u]);
      const float o = sigf(zbuf[b * COLS + 12 + u] + bias0s[12 + u]);
      const float cc = f * hp0[q] + i * g;
      const float nh = o * tanhf(cc);
      hp0[q] = nh;
      const int j = wg * UPW + u;
      p.h0bf[nxt][b * H_N + j] = (bf16)nh;
      if (t == T_N - 1) p.out[(size_t)(B_N * H_N) + b * H_N + j] = nh;
    }
    gbar(p.bar, NWG * (++epoch));   // h0' visible to all WGs

    // ---------------- Layer 1 gates: z1 = h1@Wh1^T + h0'@Wx1^T ------------
    ch = (fvec8){};
    ch = gemm_h(p.h1bf[cur], wv, Wh1s, lane, ch);
    cx = (fvec8){};
    cx = gemm_h(p.h0bf[nxt], wv, Wx1s, lane, cx);
    {
      const fvec8 z = ch + cx;
      const int colz = lane & 15;
      const int rowb = wv * 16 + ((lane & 16) ? 8 : 0);
#pragma unroll
      for (int r = 0; r < 8; ++r) zbuf[(rowb + r) * COLS + colz] = z[r];
    }
    __syncthreads();

    // cell 1
#pragma unroll
    for (int q = 0; q < 2; ++q) {
      const int pp = tid * 2 + q;
      const int b = pp >> 2, u = pp & 3;
      const float f = sigf(zbuf[b * COLS +      u] + bias1s[u]);
      const float i = sigf(zbuf[b * COLS +  4 + u] + bias1s[4 + u]);
      const float g = tanhf(zbuf[b * COLS +  8 + u] + bias1s[8 + u]);
      const float o = sigf(zbuf[b * COLS + 12 + u] + bias1s[12 + u]);
      const float cc = f * hp1[q] + i * g;
      const float nh = o * tanhf(cc);
      hp1[q] = nh;
      const int j = wg * UPW + u;
      p.h1bf[nxt][b * H_N + j] = (bf16)nh;
      if (t == T_N - 1) {
        p.out[(size_t)(2 * B_N * H_N) + b * H_N + j] = nh;
        p.last[(size_t)b * H_N + j] = nh;
      }
    }
    gbar(p.bar, NWG * (++epoch));   // h1' (and 'last') visible to all WGs
  }

  // ---------------- FC: out = last @ fc_w^T + fc_b (f32, runs once) --------
#pragma unroll
  for (int q = 0; q < 2; ++q) {
    const int pp = tid * 2 + q;
    const int b = pp >> 2, jl = pp & 3;
    const int j = wg * UPW + jl;
    float acc = p.fc_b[j];
    const float* lr = p.last + (size_t)b * H_N;
    const float* wr = p.fc_w + (size_t)j * H_N;
#pragma unroll 4
    for (int k = 0; k < H_N; ++k) acc += lr[k] * wr[k];
    p.out[(size_t)b * H_N + j] = acc;
  }
}

extern "C" void kernel_launch(void* const* d_in, const int* in_sizes, int n_in,
                              void* d_out, int out_size, void* d_ws, size_t ws_size,
                              hipStream_t stream) {
  (void)in_sizes; (void)n_in; (void)out_size; (void)ws_size;

  char* ws = (char*)d_ws;
  unsigned* bar = (unsigned*)ws;                      // 256 B reserved
  bf16* h0b0 = (bf16*)(ws + 256);
  bf16* h0b1 = h0b0 + B_N * H_N;
  bf16* h1b0 = h0b1 + B_N * H_N;
  bf16* h1b1 = h1b0 + B_N * H_N;
  float* last = (float*)(ws + 256 + (size_t)4 * B_N * H_N * sizeof(bf16));

  Params p;
  p.x      = (const float*)d_in[0];
  p.hidden = (const float*)d_in[1];
  p.fc_w   = (const float*)d_in[2];
  p.fc_b   = (const float*)d_in[3];
  for (int l = 0; l < 2; ++l)
    for (int g = 0; g < 4; ++g) {
      p.W[l * 4 + g]    = (const float*)d_in[4 + l * 8 + g * 2];
      p.bias[l * 4 + g] = (const float*)d_in[5 + l * 8 + g * 2];
    }
  p.h0bf[0] = h0b0; p.h0bf[1] = h0b1;
  p.h1bf[0] = h1b0; p.h1bf[1] = h1b1;
  p.last = last;
  p.out  = (float*)d_out;
  p.bar  = bar;

  init_kernel<<<dim3(64), dim3(256), 0, stream>>>(p.hidden, h0b0, h1b0, bar);
  lstm_kernel<<<dim3(NWG), dim3(NTHR), 0, stream>>>(p);
}
